// AngularAttention_39384850104916
// MI455X (gfx1250) — matrix operand order repaired and checked by reading, not compiled, not run on hardware
//
#include <hip/hip_runtime.h>

#define B_ 2
#define T_ 2048
#define D_ 1024
#define H_ 16
#define DK_ 64

typedef __attribute__((ext_vector_type(16))) _Float16 v16h;
typedef __attribute__((ext_vector_type(8)))  _Float16 v8h;
typedef __attribute__((ext_vector_type(8)))  float    v8f;

// ---------------------------------------------------------------------------
// WMMA / CDNA5 helpers
// ---------------------------------------------------------------------------
static __device__ __forceinline__ v8f wmma16(v16h a, v16h b, v8f c) {
  // D = A(16x32 f16) * B(32x16 f16) + C(16x16 f32)
  return __builtin_amdgcn_wmma_f32_16x16x32_f16(false, a, false, b,
                                                (short)0, c, false, false);
}

// A-fragment (16x32, f16) load, contiguous k order:
//   lane L (lo = L&15, hi = L>>4): row M = lo,
//   halves 0..7  = A[M, 2*koff+0..7]   (koff = hi*8)
//   halves 8..15 = A[M, 2*koff+8..15]
static __device__ __forceinline__ v16h load_a_frag(const _Float16* row, int koff) {
  v8h a0 = *(const v8h*)(row + 2 * koff);
  v8h a1 = *(const v8h*)(row + 2 * koff + 8);
  v16h r;
#pragma unroll
  for (int i = 0; i < 8; ++i) { r[i] = a0[i]; r[i + 8] = a1[i]; }
  return r;
}

// per-lane 16B global -> LDS async copy (gfx1250 GLOBAL_LOAD_ASYNC_TO_LDS_B128)
static __device__ __forceinline__ void async_copy_b128(const void* gaddr, void* lds) {
  unsigned l = (unsigned)(unsigned long long)lds;  // low 32 bits = LDS offset
  asm volatile("global_load_async_to_lds_b128 %0, %1, off"
               :: "v"(l), "v"(gaddr) : "memory");
}

// angular weight: clip -> 1 - acos/pi -> mask -> max(.,1e-6)^8
static __device__ __forceinline__ float angw(float sim, int mk) {
  sim = fminf(fmaxf(sim, -0.999f), 0.999f);
  float sc = 1.0f - acosf(sim) * 0.31830988618379067f;  // 1/pi
  sc = mk ? sc : 0.0f;
  float t = fmaxf(sc, 1e-6f);
  t = t * t; t = t * t; t = t * t;  // ^8
  return t;
}

// ---------------------------------------------------------------------------
// fp32 -> f16 convert
// ---------------------------------------------------------------------------
__global__ __launch_bounds__(256) void k_f32_to_f16(const float* __restrict__ s,
                                                    _Float16* __restrict__ d, int n) {
  int i = blockIdx.x * 256 + threadIdx.x;
  if (i < n) d[i] = (_Float16)s[i];
}

// ---------------------------------------------------------------------------
// GEMM epilogues (C layout: lane holds rows m0 + hi*8 + j, col n = lo)
// mode 1: per-64-col (per-head) L2-normalize rows, store f16 to Ch
// mode 2: store f16 transposed into Vt[B,H,DK,T] (Ch = Vt base)
// mode 3: add bias, store f32 to Cf
// ---------------------------------------------------------------------------
static __device__ __forceinline__ void gemm_epilogue(
    const v8f acc[4], int m0, int n0, int lo, int hi,
    float* __restrict__ Cf, _Float16* __restrict__ Ch,
    const float* __restrict__ bias, int N, int mode) {
  int r0 = hi * 8;
  if (mode == 1) {
#pragma unroll
    for (int j = 0; j < 8; ++j) {
      float ss = 0.f;
#pragma unroll
      for (int t = 0; t < 4; ++t) ss += acc[t][j] * acc[t][j];
      ss += __shfl_xor(ss, 1, 16);
      ss += __shfl_xor(ss, 2, 16);
      ss += __shfl_xor(ss, 4, 16);
      ss += __shfl_xor(ss, 8, 16);
      float inv = 1.0f / fmaxf(sqrtf(ss), 1e-12f);
#pragma unroll
      for (int t = 0; t < 4; ++t)
        Ch[(size_t)(m0 + r0 + j) * N + (n0 + t * 16 + lo)] =
            (_Float16)(acc[t][j] * inv);
    }
  } else if (mode == 2) {
#pragma unroll
    for (int j = 0; j < 8; ++j) {
      int row = m0 + r0 + j;
      int b = row >> 11, tt = row & (T_ - 1);
#pragma unroll
      for (int t = 0; t < 4; ++t) {
        int col = n0 + t * 16 + lo;
        int h = col >> 6, nl = col & 63;
        Ch[(size_t)((b * H_ + h) * DK_ + nl) * T_ + tt] = (_Float16)acc[t][j];
      }
    }
  } else {  // mode 3: bias + f32 store
#pragma unroll
    for (int j = 0; j < 8; ++j)
#pragma unroll
      for (int t = 0; t < 4; ++t) {
        int col = n0 + t * 16 + lo;
        Cf[(size_t)(m0 + r0 + j) * N + col] = acc[t][j] + bias[col];
      }
  }
}

// ---------------------------------------------------------------------------
// GEMM: C[M,N] = A[M,K] (f16, row-major) * Bt[N,K]^T (f16, "B-transposed")
// One wave computes a 32x64 tile: 2 A-fragments share all 4 B-fragments per
// k-step -> 8 WMMAs per 12 b128 loads.
// ---------------------------------------------------------------------------
__global__ __launch_bounds__(128) void k_gemm_wmma(
    const _Float16* __restrict__ A, const _Float16* __restrict__ Bt,
    float* __restrict__ Cf, _Float16* __restrict__ Ch,
    const float* __restrict__ bias, int M, int N, int K, int mode) {
  int lane = threadIdx.x & 31;
  int wave = threadIdx.x >> 5;
  int lo = lane & 15, hi = lane >> 4;
  int koff = hi * 8;

  int ntb = N >> 6;  // 64-wide col tiles
  int tile = blockIdx.x * 4 + wave;
  int tm = tile / ntb, tn = tile % ntb;
  int m0 = tm * 32, n0 = tn * 64;

  v8f acc0[4] = {};
  v8f acc1[4] = {};

  const _Float16* arow0 = A + (size_t)(m0 + lo) * K;
  const _Float16* arow1 = A + (size_t)(m0 + 16 + lo) * K;
  for (int k0 = 0; k0 < K; k0 += 32) {
    v16h a0 = load_a_frag(arow0 + k0, koff);
    v16h a1 = load_a_frag(arow1 + k0, koff);
#pragma unroll
    for (int t = 0; t < 4; ++t) {
      const _Float16* bp = Bt + (size_t)(n0 + t * 16 + lo) * K + k0 + hi * 16;
      v16h b = *(const v16h*)bp;
      acc0[t] = wmma16(a0, b, acc0[t]);
      acc1[t] = wmma16(a1, b, acc1[t]);
    }
  }

  gemm_epilogue(acc0, m0,      n0, lo, hi, Cf, Ch, bias, N, mode);
  gemm_epilogue(acc1, m0 + 16, n0, lo, hi, Cf, Ch, bias, N, mode);
}

// ---------------------------------------------------------------------------
// Angular attention. Block = 4 waves = 64 q rows of one (b,h). Per 32-key
// chunk the block stages K (32x64) and V (64x32, pre-transposed) into LDS
// with global_load_async_to_lds_b128 (ASYNCcnt), then every wave builds its
// B-fragments from LDS. P restaged through a per-wave LDS slot to convert
// C-layout -> A-fragment layout; row sums via a ones-B WMMA.
// ---------------------------------------------------------------------------
__global__ __launch_bounds__(128) void k_attn_wmma(
    const _Float16* __restrict__ Qh, const _Float16* __restrict__ Kh,
    const _Float16* __restrict__ Vt, const int* __restrict__ mask,
    _Float16* __restrict__ AO) {
  __shared__ __align__(32) _Float16 Kt[32][64];      // [key][d]   4 KB
  __shared__ __align__(32) _Float16 Vts[64][32];     // [d][key]   4 KB
  __shared__ __align__(32) _Float16 Pt[4][16][32];   // per-wave P 4 KB

  int bh = blockIdx.x;
  int b = bh >> 4, h = bh & 15;
  int wave = threadIdx.x >> 5, lane = threadIdx.x & 31;
  int lo = lane & 15, hi = lane >> 4;
  int koff = hi * 8;
  int q0 = blockIdx.y * 64 + wave * 16;
  int tid = threadIdx.x;

  // Q A-fragments for DK=64 (two k=32 chunks), held in registers all loop
  const _Float16* qrow = Qh + (size_t)(b * T_ + q0 + lo) * D_ + h * DK_;
  v16h qa0 = load_a_frag(qrow, koff);
  v16h qa1 = load_a_frag(qrow + 32, koff);

  v16h ones;
#pragma unroll
  for (int i = 0; i < 16; ++i) ones[i] = (_Float16)1.0f;

  v8f acc[4] = {};
  v8f sums = {};

  const _Float16* kbase = Kh + (size_t)(b * T_) * D_ + h * DK_;  // [T][D] slice
  const _Float16* vbase = Vt + (size_t)bh * DK_ * T_;            // [DK][T] slice

  for (int kt = 0; kt < T_; kt += 32) {
    // --- async-stage K/V chunk into LDS (256 b128 segments each -> 2/thread)
#pragma unroll
    for (int r = 0; r < 2; ++r) {
      int s = tid + r * 128;
      int krow = s >> 3, kc = (s & 7) * 8;   // K: 32 rows x 8 segs
      async_copy_b128(kbase + (size_t)(kt + krow) * D_ + kc, &Kt[krow][kc]);
      int vrow = s >> 2, vc = (s & 3) * 8;   // V: 64 rows x 4 segs
      async_copy_b128(vbase + (size_t)vrow * T_ + kt + vc, &Vts[vrow][vc]);
    }
    asm volatile("s_wait_asynccnt 0x0" ::: "memory");
    __syncthreads();

    // --- sim tiles: S0 = keys [kt, kt+16), S1 = keys [kt+16, kt+32)
    v8f s0 = {}, s1 = {};
    {
      v16h b00 = *(const v16h*)&Kt[lo][hi * 16];
      v16h b01 = *(const v16h*)&Kt[lo][32 + hi * 16];
      v16h b10 = *(const v16h*)&Kt[16 + lo][hi * 16];
      v16h b11 = *(const v16h*)&Kt[16 + lo][32 + hi * 16];
      s0 = wmma16(qa0, b00, s0);
      s0 = wmma16(qa1, b01, s0);
      s1 = wmma16(qa0, b10, s1);
      s1 = wmma16(qa1, b11, s1);
    }

    int mk0 = mask[b * T_ + kt + lo];
    int mk1 = mask[b * T_ + kt + 16 + lo];

    // --- elementwise, then restage P (C-layout -> A-fragment) via LDS
#pragma unroll
    for (int j = 0; j < 8; ++j) {
      Pt[wave][hi * 8 + j][lo]      = (_Float16)angw(s0[j], mk0);
      Pt[wave][hi * 8 + j][16 + lo] = (_Float16)angw(s1[j], mk1);
    }
    asm volatile("s_wait_dscnt 0" ::: "memory");  // in-wave DS RAW ordering
    v16h pa = load_a_frag(&Pt[wave][lo][0], koff);

    // row sums via ones-matrix WMMA (every output column = row sum)
    sums = wmma16(pa, ones, sums);

    // --- PV from LDS V tile
#pragma unroll
    for (int t = 0; t < 4; ++t) {
      v16h bv = *(const v16h*)&Vts[t * 16 + lo][hi * 16];
      acc[t] = wmma16(pa, bv, acc[t]);
    }
    __syncthreads();  // protect K/V tiles before next chunk's async stores
  }

  // --- normalize and store f16 (feeds output projection GEMM)
#pragma unroll
  for (int j = 0; j < 8; ++j) {
    float invd = 1.0f / (sums[j] + 1e-6f);
    int q = q0 + hi * 8 + j;
#pragma unroll
    for (int t = 0; t < 4; ++t)
      AO[(size_t)(b * T_ + q) * D_ + h * DK_ + t * 16 + lo] =
          (_Float16)(acc[t][j] * invd);
  }
}

// ---------------------------------------------------------------------------
// launch
// ---------------------------------------------------------------------------
extern "C" void kernel_launch(void* const* d_in, const int* in_sizes, int n_in,
                              void* d_out, int out_size, void* d_ws, size_t ws_size,
                              hipStream_t stream) {
  (void)in_sizes; (void)n_in; (void)out_size; (void)ws_size;
  const float* x  = (const float*)d_in[0];
  const int*  mask = (const int*)d_in[1];
  const float* Wq = (const float*)d_in[2];
  const float* Wk = (const float*)d_in[3];
  const float* Wv = (const float*)d_in[4];
  const float* Wo = (const float*)d_in[5];
  const float* bo = (const float*)d_in[6];
  float* out = (float*)d_out;

  char* ws = (char*)d_ws;
  _Float16* xh  = (_Float16*)(ws);                // 8 MB  : f16(x)
  _Float16* Wqh = (_Float16*)(ws + 8388608);      // 2 MB
  _Float16* Wkh = (_Float16*)(ws + 10485760);     // 2 MB
  _Float16* Wvh = (_Float16*)(ws + 12582912);     // 2 MB
  _Float16* Woh = (_Float16*)(ws + 14680064);     // 2 MB
  _Float16* Qh  = (_Float16*)(ws + 16777216);     // 8 MB  : normalized Q
  _Float16* Kh  = (_Float16*)(ws + 25165824);     // 8 MB  : normalized K
  _Float16* Vt  = (_Float16*)(ws + 33554432);     // 8 MB  : V transposed [B,H,DK,T]
  _Float16* AO  = (_Float16*)(ws + 41943040);     // 8 MB  : attention output

  const int nx = B_ * T_ * D_;  // 4194304
  const int nw = D_ * D_;       // 1048576
  k_f32_to_f16<<<nx / 256, 256, 0, stream>>>(x,  xh,  nx);
  k_f32_to_f16<<<nw / 256, 256, 0, stream>>>(Wq, Wqh, nw);
  k_f32_to_f16<<<nw / 256, 256, 0, stream>>>(Wk, Wkh, nw);
  k_f32_to_f16<<<nw / 256, 256, 0, stream>>>(Wv, Wvh, nw);
  k_f32_to_f16<<<nw / 256, 256, 0, stream>>>(Wo, Woh, nw);

  const int M = B_ * T_, N = D_, K = D_;
  const int blocks = (M / 32) * (N / 64) / 4;  // 4 waves/block, 32x64 per wave
  k_gemm_wmma<<<blocks, 128, 0, stream>>>(xh, Wqh, nullptr, Qh, nullptr, M, N, K, 1);
  k_gemm_wmma<<<blocks, 128, 0, stream>>>(xh, Wkh, nullptr, Kh, nullptr, M, N, K, 1);
  k_gemm_wmma<<<blocks, 128, 0, stream>>>(xh, Wvh, nullptr, Vt, nullptr, M, N, K, 2);

  k_attn_wmma<<<dim3(B_ * H_, T_ / 64), 128, 0, stream>>>(Qh, Kh, Vt, mask, AO);

  k_gemm_wmma<<<blocks, 128, 0, stream>>>(AO, Woh, out, nullptr, bo, M, N, K, 3);
}
